// DftSpectrogram_80075370267252
// MI455X (gfx1250) — matrix operand order repaired and checked by reading, not compiled, not run on hardware
//
#include <hip/hip_runtime.h>
#include <math.h>

typedef __attribute__((ext_vector_type(16))) _Float16 v16h;
typedef __attribute__((ext_vector_type(8)))  _Float16 v8h;
typedef __attribute__((ext_vector_type(8)))  float    v8f;

#define NFFT      512
#define SHIFT     160
#define NFRAMES   601
#define KOUT      256
#define TLEN      96512
#define NBATCH    32
#define TILE      64
#define KSTEP     32
#define LDSS      40          // 32 halves + 8 pad (80B row stride, 16B aligned)
#define EPSF      1e-7f
#define HALF_INV_LN10 0.21714724095162590f   // 0.5 / ln(10)

static __device__ inline v16h concat8(v8h lo, v8h hi) {
  return __builtin_shufflevector(lo, hi, 0,1,2,3,4,5,6,7,8,9,10,11,12,13,14,15);
}

__global__ __launch_bounds__(256) void dft_gemm_logmag_kernel(
    const float* __restrict__ x, const float* __restrict__ Wr,
    const float* __restrict__ Wi, float* __restrict__ out)
{
  __shared__ __align__(16) _Float16 sWr[TILE * LDSS];
  __shared__ __align__(16) _Float16 sWi[TILE * LDSS];
  __shared__ __align__(16) _Float16 sF [TILE * LDSS];

  const int tid   = threadIdx.x;
  const int lane  = tid & 31;
  const int wave  = tid >> 5;      // 0..7
  const int waveM = wave >> 1;     // 0..3  -> 16 k-rows each
  const int waveN = wave & 1;      // 0..1  -> 32 f-cols each

  const int ft = blockIdx.x;       // 0..9   frame tile
  const int kt = blockIdx.y;       // 0..3   freq tile
  const int b  = blockIdx.z;       // 0..31  batch

  const int k0 = kt * TILE;
  const int f0 = ft * TILE;
  const float* xb = x + (size_t)b * TLEN;

  // cooperative-load mapping: 4 threads per row, 8 consecutive floats each
  const int lrow = tid >> 2;          // 0..63
  const int lcol = (tid & 3) * 8;     // 0,8,16,24
  const int fRow = f0 + lrow;
  const bool fOK = (fRow < NFRAMES);

  v8f cr0 = {}, cr1 = {}, ci0 = {}, ci1 = {};

  for (int kk = 0; kk < NFFT; kk += KSTEP) {
    // ---- stage tiles into LDS with f32 -> f16 conversion ----
    {
      const float* wrp = Wr + (size_t)(k0 + lrow) * NFFT + kk + lcol;
      const float* wip = Wi + (size_t)(k0 + lrow) * NFFT + kk + lcol;
      _Float16* dWr = &sWr[lrow * LDSS + lcol];
      _Float16* dWi = &sWi[lrow * LDSS + lcol];
      _Float16* dF  = &sF [lrow * LDSS + lcol];
#pragma unroll
      for (int i = 0; i < 8; ++i) {
        dWr[i] = (_Float16)wrp[i];
        dWi[i] = (_Float16)wip[i];
      }
      if (fOK) {
        const float* fp = xb + (size_t)fRow * SHIFT + kk + lcol;
#pragma unroll
        for (int i = 0; i < 8; ++i) dF[i] = (_Float16)fp[i];
      } else {
#pragma unroll
        for (int i = 0; i < 8; ++i) dF[i] = (_Float16)0.0f;
      }
    }
    __syncthreads();

    // ---- A fragments (16x32 f16): lanes 0-15 hold K{0..7,16..23},
    //      lanes 16-31 hold K{8..15,24..31} of row M = lane%16 ----
    const int offA = (lane < 16) ? 0 : 8;
    const int mrow = waveM * 16 + (lane & 15);
    const _Float16* arp = &sWr[mrow * LDSS + offA];
    const _Float16* aip = &sWi[mrow * LDSS + offA];
    v16h Ar = concat8(*(const v8h*)(arp), *(const v8h*)(arp + 16));
    v16h Ai = concat8(*(const v8h*)(aip), *(const v8h*)(aip + 16));

    // ---- B fragments (32x16 f16): lane holds column N = lane%16 with
    //      lanes 0-15: K{0..7,16..23}, lanes 16-31: K{8..15,24..31} ----
    const int offB  = (lane < 16) ? 0 : 8;
    const int nrow0 = waveN * 32 + (lane & 15);
    const _Float16* b0p = &sF[nrow0        * LDSS + offB];
    const _Float16* b1p = &sF[(nrow0 + 16) * LDSS + offB];
    v16h B0 = concat8(*(const v8h*)(b0p), *(const v8h*)(b0p + 16));
    v16h B1 = concat8(*(const v8h*)(b1p), *(const v8h*)(b1p + 16));

    cr0 = __builtin_amdgcn_wmma_f32_16x16x32_f16(false, Ar, false, B0, (short)0, cr0, false, false);
    cr1 = __builtin_amdgcn_wmma_f32_16x16x32_f16(false, Ar, false, B1, (short)0, cr1, false, false);
    ci0 = __builtin_amdgcn_wmma_f32_16x16x32_f16(false, Ai, false, B0, (short)0, ci0, false, false);
    ci1 = __builtin_amdgcn_wmma_f32_16x16x32_f16(false, Ai, false, B1, (short)0, ci1, false, false);

    __syncthreads();
  }

  // ---- epilogue: C/D layout: lane N = lane%16; VGPR r -> M = r + (lane<16?0:8) ----
  const int fIdx0 = f0 + waveN * 32 + (lane & 15);
  const int kBase = k0 + waveM * 16 + ((lane < 16) ? 0 : 8);
#pragma unroll
  for (int r = 0; r < 8; ++r) {
    const int kIdx = kBase + r;
    float* orow = out + ((size_t)b * KOUT + kIdx) * NFRAMES;
    float re = cr0[r], im = ci0[r];
    float v  = HALF_INV_LN10 * logf(re * re + im * im + EPSF);
    if (fIdx0 < NFRAMES) orow[fIdx0] = v;
    re = cr1[r]; im = ci1[r];
    v  = HALF_INV_LN10 * logf(re * re + im * im + EPSF);
    if (fIdx0 + 16 < NFRAMES) orow[fIdx0 + 16] = v;
  }
}

// Per-(b,f) column: mean/std over 256 bins, normalize in place. One wave per column.
__global__ __launch_bounds__(256) void norm_kernel(float* __restrict__ out)
{
  const int lane = threadIdx.x & 31;
  const int wave = threadIdx.x >> 5;
  const int col  = blockIdx.x * 8 + wave;       // 0 .. 32*601-1
  if (col >= NBATCH * NFRAMES) return;          // uniform per wave
  const int b = col / NFRAMES;
  const int f = col - b * NFRAMES;
  float* base = out + (size_t)b * KOUT * NFRAMES + f;

  float v[8];
  float s = 0.f, s2 = 0.f;
#pragma unroll
  for (int i = 0; i < 8; ++i) {
    v[i] = base[(size_t)(lane + 32 * i) * NFRAMES];
    s  += v[i];
    s2 += v[i] * v[i];
  }
#pragma unroll
  for (int off = 16; off > 0; off >>= 1) {
    s  += __shfl_xor(s,  off, 32);
    s2 += __shfl_xor(s2, off, 32);
  }
  const float mean = s * (1.0f / 256.0f);
  const float var  = fmaxf(s2 * (1.0f / 256.0f) - mean * mean, 0.0f);
  const float inv  = 1.0f / (sqrtf(var) + EPSF);
#pragma unroll
  for (int i = 0; i < 8; ++i)
    base[(size_t)(lane + 32 * i) * NFRAMES] = (v[i] - mean) * inv;
}

extern "C" void kernel_launch(void* const* d_in, const int* in_sizes, int n_in,
                              void* d_out, int out_size, void* d_ws, size_t ws_size,
                              hipStream_t stream) {
  (void)in_sizes; (void)n_in; (void)out_size; (void)d_ws; (void)ws_size;
  const float* x  = (const float*)d_in[0];
  const float* Wr = (const float*)d_in[1];
  const float* Wi = (const float*)d_in[2];
  float* out = (float*)d_out;

  dim3 grid((NFRAMES + TILE - 1) / TILE, KOUT / TILE, NBATCH);  // 10 x 4 x 32
  dft_gemm_logmag_kernel<<<grid, 256, 0, stream>>>(x, Wr, Wi, out);

  const int ncols = NBATCH * NFRAMES;            // 19232, divisible by 8
  norm_kernel<<<dim3(ncols / 8), 256, 0, stream>>>(out);
}
